// PixelAggregationNetwork_60833916780679
// MI455X (gfx1250) — compile-verified, matrix operand order checked by reading, not run on hardware
//
#include <hip/hip_runtime.h>
#include <hip/hip_bf16.h>
#include <math.h>

// ---------------- problem constants (from the reference) ----------------
#define BATCH 4
#define CHN   32
#define HH    512
#define WW    500
#define SEG   10
#define SWID  (WW / SEG)          // 50
#define GCH   32                  // row-chunks per (b,s) tile
#define ROWS  (HH / GCH)          // 16 rows per chunk
#define PIX   (ROWS * SWID)       // 800 pixels per chunk
#define PIX2  (PIX / 2)           // 400 float2 groups per chunk
#define SWID2 (SWID / 2)          // 25 float2 per row
#define NPART 33                  // 32 channel sums + 1 graySq sum

typedef __attribute__((ext_vector_type(16))) _Float16 v16h;
typedef __attribute__((ext_vector_type(8)))  float    v8f;

__device__ __forceinline__ float wave_sum32(float v) {
#pragma unroll
  for (int m = 16; m >= 1; m >>= 1) v += __shfl_xor(v, m, 32);
  return v;
}

// ---------------------------------------------------------------------------
// Kernel 1: single pass over x (131 MB -> HBM-bound, ~5.6us at 23.3 TB/s).
// grid = (GCH, BATCH*SEG) blocks of 256 threads -> 1280 blocks (~10K waves).
// float2 loads (all address components even -> 8B aligned), per-thread
// 32-channel accumulators + (channel-sum)^2 for the texture term.
// Cross-lane channel reduction: log-fold transpose-reduce, 31 shuffles total
// (vs 160 for the naive per-channel butterfly). Lane c ends with channel c.
// ---------------------------------------------------------------------------
__global__ __launch_bounds__(256)
void pan_reduce_kernel(const float* __restrict__ x, float* __restrict__ partials) {
  const int tile  = blockIdx.y;          // b*SEG + s
  const int b     = tile / SEG;
  const int s     = tile % SEG;
  const int chunk = blockIdx.x;
  const int tid   = threadIdx.x;
  const int lane  = tid & 31;
  const int wave  = tid >> 5;

  float chan[CHN];
#pragma unroll
  for (int c = 0; c < CHN; ++c) chan[c] = 0.0f;
  float gsq = 0.0f;

  const size_t plane = (size_t)HH * WW;  // channel stride (elements)

  for (int p2 = tid; p2 < PIX2; p2 += 256) {
    const int r   = p2 / SWID2;
    const int wl2 = p2 - r * SWID2;
    const int h   = chunk * ROWS + r;
    const int w   = s * SWID + wl2 * 2;
    const float* base = x + (size_t)b * CHN * plane + (size_t)h * WW + w;
    float psx = 0.0f, psy = 0.0f;
#pragma unroll
    for (int c = 0; c < CHN; ++c) {
      float2 v = *reinterpret_cast<const float2*>(base + (size_t)c * plane);
      psx += v.x;
      psy += v.y;
      chan[c] += v.x + v.y;
    }
    gsq += psx * psx + psy * psy;
  }

  // log-fold transpose-reduce: after step mask m, each lane tracks only the
  // channels whose bit (m) matches its own lane bit; v[0] ends as the wave
  // total of channel == lane.  16+8+4+2+1 = 31 shuffles.
#pragma unroll
  for (int m = 16; m >= 1; m >>= 1) {
    const bool up = (lane & m) != 0;
#pragma unroll
    for (int i = 0; i < m; ++i) {
      float send = up ? chan[i] : chan[i + m];
      float keep = up ? chan[i + m] : chan[i];
      chan[i] = keep + __shfl_xor(send, m, 32);
    }
  }
  const float myChan = chan[0];
  const float g = wave_sum32(gsq);

  __shared__ float lds[8 * 32 + 8];
  lds[wave * 32 + lane] = myChan;
  if (lane == 0) lds[256 + wave] = g;
  __syncthreads();

  if (tid < 32) {
    float sum = 0.0f;
#pragma unroll
    for (int wv = 0; wv < 8; ++wv) sum += lds[wv * 32 + tid];
    partials[((size_t)tile * GCH + chunk) * NPART + tid] = sum;
  }
  if (tid == 0) {
    float sg = 0.0f;
#pragma unroll
    for (int wv = 0; wv < 8; ++wv) sg += lds[256 + wv];
    partials[((size_t)tile * GCH + chunk) * NPART + 32] = sg;
  }
}

// ---------------------------------------------------------------------------
// Kernel 2: fold chunk-partials per (b,s) into feats[b][s][c].
// nodes = chanSum/25600 ; texture = graySqSum/1024 (gray = pixSum/32)
// feats = 0.5*nodes + 0.5*texture
// ---------------------------------------------------------------------------
__global__ __launch_bounds__(64)
void pan_feats_kernel(const float* __restrict__ partials, float* __restrict__ feats) {
  const int tile = blockIdx.x;
  const int t    = threadIdx.x;
  __shared__ float stex;

  float sum = 0.0f;
  if (t < NPART) {
    for (int k = 0; k < GCH; ++k)
      sum += partials[((size_t)tile * GCH + k) * NPART + t];
    if (t == 32) stex = sum * (1.0f / 1024.0f);   // texture[b,s]
  }
  __syncthreads();
  if (t < 32) {
    float node = sum * (1.0f / (float)(HH * SWID)); // /25600
    feats[(size_t)tile * 32 + t] = 0.5f * node + 0.5f * stex;
  }
}

// ---------------------------------------------------------------------------
// Kernel 3: single wave (32 lanes == 32 channels, EXEC all ones for WMMA).
//  - hierarchy reg term (level sizes 10->5->3->2->1, mean-of-pairs)
//  - center feats across segments (distance-invariant; fixes f16 cancellation)
//  - Gram = F * F^T via one v_wmma_f32_16x16x32_f16 per batch (K=32=C)
//  - contrastive loss from d^2 = Gii + Gjj - 2Gij, w = 1 - LCAlevel/4
//  - constant tree output + scalar loss
// fc is zero-padded to 16 rows so the A-pack is branch-free (ds_load_b128).
// ---------------------------------------------------------------------------
__global__ __launch_bounds__(32)
void pan_loss_kernel(const float* __restrict__ feats, float* __restrict__ out) {
  const int lane = threadIdx.x;          // == channel
  __shared__ float fc[16 * 32];          // centered feats, [row<=15][c], rows>=10 zero
  __shared__ float gram[16 * 16];

  float regP = 0.0f, lossP = 0.0f;

  for (int b = 0; b < BATCH; ++b) {
    float f[SEG];
#pragma unroll
    for (int s = 0; s < SEG; ++s) f[s] = feats[(size_t)(b * SEG + s) * 32 + lane];

    // ---- hierarchical mean-of-pairs + reg = sum of per-level mean(cur^2) ----
    float l1[5], l2[3], l3[2], l4;
#pragma unroll
    for (int i = 0; i < 5; ++i) l1[i] = 0.5f * (f[2 * i] + f[2 * i + 1]);
    l2[0] = 0.5f * (l1[0] + l1[1]); l2[1] = 0.5f * (l1[2] + l1[3]); l2[2] = l1[4];
    l3[0] = 0.5f * (l2[0] + l2[1]); l3[1] = l2[2];
    l4    = 0.5f * (l3[0] + l3[1]);
    float s1 = 0.0f, s2 = 0.0f, s3 = 0.0f;
#pragma unroll
    for (int i = 0; i < 5; ++i) s1 += l1[i] * l1[i];
#pragma unroll
    for (int i = 0; i < 3; ++i) s2 += l2[i] * l2[i];
    s3 = l3[0] * l3[0] + l3[1] * l3[1];
    regP += s1 * (1.0f / (BATCH * 5 * 32))
          + s2 * (1.0f / (BATCH * 3 * 32))
          + s3 * (1.0f / (BATCH * 2 * 32))
          + l4 * l4 * (1.0f / (BATCH * 1 * 32));

    // ---- center across segments (pairwise distances unchanged) ----
    float m = 0.0f;
#pragma unroll
    for (int s = 0; s < SEG; ++s) m += f[s];
    m *= 0.1f;
#pragma unroll
    for (int s = 0; s < SEG; ++s) fc[s * 32 + lane] = f[s] - m;
#pragma unroll
    for (int s = SEG; s < 16; ++s) fc[s * 32 + lane] = 0.0f;   // zero-pad rows
    __syncthreads();

    // ---- pack A (16x32 f16). ISA layout:
    //   lanes 0-15 : M=lane,    elems 0..7 -> K=0..7,  elems 8..15 -> K=16..23
    //   lanes16-31 : M=lane-16, elems 0..7 -> K=8..15, elems 8..15 -> K=24..31
    // B = A^T has the mirrored striping -> same register contents.
    const int M     = lane & 15;
    const int kbase = (lane < 16) ? 0 : 8;
    const float* row = &fc[M * 32 + kbase];
    float4 q0 = *reinterpret_cast<const float4*>(row + 0);
    float4 q1 = *reinterpret_cast<const float4*>(row + 4);
    float4 q2 = *reinterpret_cast<const float4*>(row + 16);
    float4 q3 = *reinterpret_cast<const float4*>(row + 20);
    v16h a;
    a[0]  = (_Float16)q0.x; a[1]  = (_Float16)q0.y;
    a[2]  = (_Float16)q0.z; a[3]  = (_Float16)q0.w;
    a[4]  = (_Float16)q1.x; a[5]  = (_Float16)q1.y;
    a[6]  = (_Float16)q1.z; a[7]  = (_Float16)q1.w;
    a[8]  = (_Float16)q2.x; a[9]  = (_Float16)q2.y;
    a[10] = (_Float16)q2.z; a[11] = (_Float16)q2.w;
    a[12] = (_Float16)q3.x; a[13] = (_Float16)q3.y;
    a[14] = (_Float16)q3.z; a[15] = (_Float16)q3.w;

    v8f acc = {};
    acc = __builtin_amdgcn_wmma_f32_16x16x32_f16(
        /*neg_a=*/false, a, /*neg_b=*/false, a,
        /*c_mod=*/(short)0, acc, /*reuse_a=*/false, /*reuse_b=*/false);

    // ---- scatter Gram to LDS (C/D layout: vgpr r -> M=r(+8), N=lane&15) ----
    const int mofs = (lane < 16) ? 0 : 8;
#pragma unroll
    for (int r = 0; r < 8; ++r) gram[(r + mofs) * 16 + (lane & 15)] = acc[r];
    __syncthreads();

    // ---- contrastive terms over all 10x10 pairs ----
    for (int p = lane; p < SEG * SEG; p += 32) {
      int i = p / SEG, j = p % SEG;
      float d2 = gram[i * 16 + i] + gram[j * 16 + j] - 2.0f * gram[i * 16 + j];
      d2 = fmaxf(d2, 0.0f) + 1e-12f;
      float d = sqrtf(d2);
      int ai = i, aj = j, l = 0;
      while (ai != aj) { ai >>= 1; aj >>= 1; ++l; }   // LCA level (parent = i/2)
      float w = 1.0f - 0.25f * (float)l;              // MAXL = 4
      float hg = fmaxf(1.0f - d, 0.0f);               // MARGIN = 1
      lossP += w * d2 + (1.0f - w) * hg * hg;
    }
    __syncthreads();   // before fc/gram reuse next batch
  }

  lossP = wave_sum32(lossP);
  regP  = wave_sum32(regP);
  if (lane == 0)
    out[BATCH * 21 * 2] = lossP * (1.0f / (BATCH * SEG * SEG)) + 0.01f * regP;

  // ---- constant tree output: [B,21,2] = (parent, level), broadcast over B ----
  if (lane < 21) {
    const int offs[6] = {0, 10, 15, 18, 20, 21};
    int gidx = lane, lvl = 0;
    while (gidx >= offs[lvl + 1]) ++lvl;
    int i = gidx - offs[lvl];
    int parent = (lvl < 4) ? (offs[lvl + 1] + (i >> 1)) : gidx;
#pragma unroll
    for (int b = 0; b < BATCH; ++b) {
      out[b * 42 + gidx * 2 + 0] = (float)parent;
      out[b * 42 + gidx * 2 + 1] = (float)lvl;
    }
  }
}

// ---------------------------------------------------------------------------
extern "C" void kernel_launch(void* const* d_in, const int* in_sizes, int n_in,
                              void* d_out, int out_size, void* d_ws, size_t ws_size,
                              hipStream_t stream) {
  (void)in_sizes; (void)n_in; (void)out_size; (void)ws_size;
  const float* x = (const float*)d_in[0];
  float* out = (float*)d_out;
  float* ws  = (float*)d_ws;

  float* partials = ws;                                     // 40*32*33 floats
  float* feats    = ws + (size_t)BATCH * SEG * GCH * NPART; // 4*10*32 floats

  dim3 g1(GCH, BATCH * SEG);
  pan_reduce_kernel<<<g1, 256, 0, stream>>>(x, partials);
  pan_feats_kernel<<<BATCH * SEG, 64, 0, stream>>>(partials, feats);
  pan_loss_kernel<<<1, 32, 0, stream>>>(feats, out);
}
